// HSTULiteInteraction_14834817040553
// MI455X (gfx1250) — matrix operand order repaired and checked by reading, not compiled, not run on hardware
//
#include <hip/hip_runtime.h>
#include <hip/hip_bf16.h>

#define BATCH 2048
#define SEQ   39
#define DM    256
#define NH    8
#define NL    4
#define FF    1024
#define NDD   13
#define NSS   26
#define VOC   1001
#define HDIM  32
#define TOK   (BATCH*SEQ)

typedef unsigned short u16;
typedef __attribute__((ext_vector_type(16))) __bf16 v16bf;
typedef __attribute__((ext_vector_type(8)))  float  v8f;

union FragBF {
    v16bf v;
    int4  i[2];
};

static __device__ __forceinline__ u16 f2bf(float f) {
    unsigned int u = __float_as_uint(f);
    u += 0x7FFFu + ((u >> 16) & 1u);          // round-to-nearest-even
    return (u16)(u >> 16);
}

// ---------------------------------------------------------------------------
// fp32 -> bf16 elementwise convert
// ---------------------------------------------------------------------------
__global__ __launch_bounds__(256) void k_f32_to_bf16(const float* __restrict__ in,
                                                     u16* __restrict__ out, size_t n) {
    size_t i = (size_t)blockIdx.x * 256 + threadIdx.x;
    if (i < n) out[i] = f2bf(in[i]);
}

// ---------------------------------------------------------------------------
// Embedding: dense (B,13)->(B,13,256) matvec + sparse gathers -> x (B,39,256)
// ---------------------------------------------------------------------------
__global__ __launch_bounds__(256) void k_embed(const float* __restrict__ dense_x,
                                               const int*   __restrict__ sparse_x,
                                               const float* __restrict__ dense_w,
                                               const float* __restrict__ dense_b,
                                               const float* __restrict__ emb,
                                               float* __restrict__ x) {
    size_t idx = (size_t)blockIdx.x * 256 + threadIdx.x;
    if (idx >= (size_t)TOK * DM) return;
    int d = (int)(idx % DM);
    int s = (int)((idx / DM) % SEQ);
    int b = (int)(idx / ((size_t)DM * SEQ));
    float val;
    if (s < NDD) {
        int j = s * DM + d;
        float acc = dense_b[j];
        const float* wr = dense_w + (size_t)j * NDD;
        const float* xr = dense_x + (size_t)b * NDD;
#pragma unroll
        for (int i = 0; i < NDD; ++i) acc += xr[i] * wr[i];
        val = acc;
    } else {
        int t = s - NDD;
        int row = sparse_x[(size_t)b * NSS + t];
        val = emb[((size_t)t * VOC + row) * DM + d];
    }
    x[idx] = val;
}

// ---------------------------------------------------------------------------
// LayerNorm over D=256, one token per block, bf16 output
// ---------------------------------------------------------------------------
__global__ __launch_bounds__(256) void k_layernorm_bf16(const float* __restrict__ x,
                                                        const float* __restrict__ g,
                                                        const float* __restrict__ b,
                                                        u16* __restrict__ out) {
    int t = blockIdx.x;
    int tid = threadIdx.x;
    const float* xr = x + (size_t)t * DM;
    float v = xr[tid];
    __shared__ float sb[256];
    sb[tid] = v; __syncthreads();
    for (int st = 128; st > 0; st >>= 1) { if (tid < st) sb[tid] += sb[tid + st]; __syncthreads(); }
    float mean = sb[0] * (1.0f / DM);
    __syncthreads();
    float c = v - mean;
    sb[tid] = c * c; __syncthreads();
    for (int st = 128; st > 0; st >>= 1) { if (tid < st) sb[tid] += sb[tid + st]; __syncthreads(); }
    float var = sb[0] * (1.0f / DM);
    float r = c * rsqrtf(var + 1e-5f) * g[tid] + b[tid];
    out[(size_t)t * DM + tid] = f2bf(r);
}

// ---------------------------------------------------------------------------
// WMMA bf16 GEMM:  C(M,N) = A(M,K) @ B(N,K)^T   (+bias, optional relu/resid)
// 256 threads = 8 waves; each wave owns a 32x64 tile (2 M-subtiles x 4 N-subtiles);
// K stepped by 32. All 12 loads of a K-step are issued before its 8 WMMAs so the
// matrix pipe runs back-to-back with a single loadcnt wait per iteration.
// ---------------------------------------------------------------------------
template <int RELU, int RESID, int OUTBF>
__global__ __launch_bounds__(256) void k_gemm_wmma(const u16* __restrict__ A,
                                                   const u16* __restrict__ Bw,
                                                   const float* __restrict__ bias,
                                                   float* __restrict__ Cf,
                                                   u16* __restrict__ Cbf,
                                                   const float* __restrict__ resid,
                                                   int M, int N, int K) {
    int wave = threadIdx.x >> 5;
    int lane = threadIdx.x & 31;
    int m0 = (blockIdx.x * 8 + wave) * 32;
    int n0 = blockIdx.y * 64;
    int r  = lane & 15;
    int hi = lane >> 4;

    // A fragment rows: lane<16 holds K {0..7}+{16..23}, lane>=16 holds {8..15}+{24..31}
    const u16* Arow0 = A + (size_t)(m0 + r) * K + hi * 8;
    const u16* Arow1 = Arow0 + (size_t)16 * K;
    // B fragment rows: lane<16 holds K {0..7}+{16..23}, lane>=16 holds {8..15}+{24..31}
    const u16* Brow = Bw + (size_t)(n0 + r) * K + hi * 8;

    v8f acc[2][4] = {{v8f{0}, v8f{0}, v8f{0}, v8f{0}},
                     {v8f{0}, v8f{0}, v8f{0}, v8f{0}}};

    for (int k0 = 0; k0 < K; k0 += 32) {
        __builtin_prefetch(Arow0 + k0 + 256, 0, 3);
        __builtin_prefetch(Arow1 + k0 + 256, 0, 3);
        FragBF a0, a1, bf[4];
        // batch all loads first: 4 A dwords4 + 8 B dwords4
        a0.i[0] = *(const int4*)(Arow0 + k0);
        a0.i[1] = *(const int4*)(Arow0 + k0 + 16);
        a1.i[0] = *(const int4*)(Arow1 + k0);
        a1.i[1] = *(const int4*)(Arow1 + k0 + 16);
#pragma unroll
        for (int t = 0; t < 4; ++t) {
            const u16* bp = Brow + (size_t)t * 16 * K + k0;
            bf[t].i[0] = *(const int4*)(bp);
            bf[t].i[1] = *(const int4*)(bp + 16);
        }
#pragma unroll
        for (int t = 0; t < 4; ++t) {
            acc[0][t] = __builtin_amdgcn_wmma_f32_16x16x32_bf16(
                false, a0.v, false, bf[t].v, (short)0, acc[0][t], false, false);
            acc[1][t] = __builtin_amdgcn_wmma_f32_16x16x32_bf16(
                false, a1.v, false, bf[t].v, (short)0, acc[1][t], false, false);
        }
    }

    // D layout: vgpr rr -> M = mtile + rr + (lane>=16 ? 8 : 0), N = n0 + t*16 + (lane&15)
    int col = r;
#pragma unroll
    for (int s = 0; s < 2; ++s) {
        int mbase = m0 + s * 16 + hi * 8;
#pragma unroll
        for (int t = 0; t < 4; ++t) {
            int n = n0 + t * 16 + col;
            float bv = bias[n];
#pragma unroll
            for (int rr = 0; rr < 8; ++rr) {
                float val = acc[s][t][rr] + bv;
                if (RELU) val = fmaxf(val, 0.0f);
                size_t off = (size_t)(mbase + rr) * N + n;
                if (RESID) val += resid[off];
                if (OUTBF) Cbf[off] = f2bf(val);
                else       Cf[off]  = val;
            }
        }
    }
}

// ---------------------------------------------------------------------------
// Sigmoid attention (per layer): one block per (batch, head); x += o
// q/k/v are (TOK, 256) fp32, head h at column h*32
// ---------------------------------------------------------------------------
__global__ __launch_bounds__(256) void k_attn_sigmoid(const float* __restrict__ q,
                                                      const float* __restrict__ k,
                                                      const float* __restrict__ v,
                                                      float* __restrict__ x,
                                                      float scale) {
    int b = blockIdx.x, h = blockIdx.y;
    __shared__ float sq[SEQ][HDIM + 1], sk[SEQ][HDIM + 1], sv[SEQ][HDIM + 1];
    __shared__ float sa[SEQ][SEQ + 1];
    size_t base = ((size_t)b * SEQ) * DM + h * HDIM;
    for (int idx = threadIdx.x; idx < SEQ * HDIM; idx += 256) {
        int s = idx >> 5, d = idx & 31;
        size_t off = base + (size_t)s * DM + d;
        sq[s][d] = q[off]; sk[s][d] = k[off]; sv[s][d] = v[off];
    }
    __syncthreads();
    for (int idx = threadIdx.x; idx < SEQ * SEQ; idx += 256) {
        int qi = idx / SEQ, ki = idx % SEQ;
        float acc = 0.f;
#pragma unroll
        for (int d = 0; d < HDIM; ++d) acc += sq[qi][d] * sk[ki][d];
        sa[qi][ki] = 1.0f / (1.0f + expf(-acc * scale));
    }
    __syncthreads();
    for (int idx = threadIdx.x; idx < SEQ * HDIM; idx += 256) {
        int qi = idx >> 5, d = idx & 31;
        float acc = 0.f;
        for (int ki = 0; ki < SEQ; ++ki) acc += sa[qi][ki] * sv[ki][d];
        x[base + (size_t)qi * DM + d] += acc;
    }
}

// ---------------------------------------------------------------------------
// Softmax attention (final MHA): writes o buffer
// ---------------------------------------------------------------------------
__global__ __launch_bounds__(256) void k_attn_softmax(const float* __restrict__ q,
                                                      const float* __restrict__ k,
                                                      const float* __restrict__ v,
                                                      float* __restrict__ o,
                                                      float scale) {
    int b = blockIdx.x, h = blockIdx.y;
    __shared__ float sq[SEQ][HDIM + 1], sk[SEQ][HDIM + 1], sv[SEQ][HDIM + 1];
    __shared__ float sa[SEQ][SEQ + 1];
    size_t base = ((size_t)b * SEQ) * DM + h * HDIM;
    for (int idx = threadIdx.x; idx < SEQ * HDIM; idx += 256) {
        int s = idx >> 5, d = idx & 31;
        size_t off = base + (size_t)s * DM + d;
        sq[s][d] = q[off]; sk[s][d] = k[off]; sv[s][d] = v[off];
    }
    __syncthreads();
    for (int idx = threadIdx.x; idx < SEQ * SEQ; idx += 256) {
        int qi = idx / SEQ, ki = idx % SEQ;
        float acc = 0.f;
#pragma unroll
        for (int d = 0; d < HDIM; ++d) acc += sq[qi][d] * sk[ki][d];
        sa[qi][ki] = acc * scale;
    }
    __syncthreads();
    if (threadIdx.x < SEQ) {
        int qi = threadIdx.x;
        float m = -3.4e38f;
        for (int ki = 0; ki < SEQ; ++ki) m = fmaxf(m, sa[qi][ki]);
        float s = 0.f;
        for (int ki = 0; ki < SEQ; ++ki) { float e = expf(sa[qi][ki] - m); sa[qi][ki] = e; s += e; }
        float inv = 1.0f / s;
        for (int ki = 0; ki < SEQ; ++ki) sa[qi][ki] *= inv;
    }
    __syncthreads();
    for (int idx = threadIdx.x; idx < SEQ * HDIM; idx += 256) {
        int qi = idx >> 5, d = idx & 31;
        float acc = 0.f;
        for (int ki = 0; ki < SEQ; ++ki) acc += sa[qi][ki] * sv[ki][d];
        o[base + (size_t)qi * DM + d] = acc;
    }
}

// ---------------------------------------------------------------------------
// Final: out[b] = sum_j g[b,j] * w2[j] + b2
// ---------------------------------------------------------------------------
__global__ __launch_bounds__(256) void k_final_reduce(const float* __restrict__ g,
                                                      const float* __restrict__ w2,
                                                      const float* __restrict__ b2,
                                                      float* __restrict__ out) {
    int b = blockIdx.x, tid = threadIdx.x;
    __shared__ float sb[256];
    const float* gr = g + (size_t)b * FF;
    float s = 0.f;
    for (int j = tid; j < FF; j += 256) s += gr[j] * w2[j];
    sb[tid] = s; __syncthreads();
    for (int st = 128; st > 0; st >>= 1) { if (tid < st) sb[tid] += sb[tid + st]; __syncthreads(); }
    if (tid == 0) out[b] = sb[0] + b2[0];
}

// ---------------------------------------------------------------------------
extern "C" void kernel_launch(void* const* d_in, const int* in_sizes, int n_in,
                              void* d_out, int out_size, void* d_ws, size_t ws_size,
                              hipStream_t stream) {
    const float* dense_x = (const float*)d_in[0];
    const int*   sparse_x= (const int*)  d_in[1];
    const float* dense_w = (const float*)d_in[2];
    const float* dense_b = (const float*)d_in[3];
    const float* emb     = (const float*)d_in[4];
    const float* Wq  = (const float*)d_in[5];
    const float* bq  = (const float*)d_in[6];
    const float* Wk  = (const float*)d_in[7];
    const float* bk  = (const float*)d_in[8];
    const float* Wv  = (const float*)d_in[9];
    const float* bv  = (const float*)d_in[10];
    const float* W1f = (const float*)d_in[11];
    const float* b1f = (const float*)d_in[12];
    const float* W2f = (const float*)d_in[13];
    const float* b2f = (const float*)d_in[14];
    const float* ln1_g = (const float*)d_in[15];
    const float* ln1_b = (const float*)d_in[16];
    const float* ln2_g = (const float*)d_in[17];
    const float* ln2_b = (const float*)d_in[18];
    const float* min_w = (const float*)d_in[19];
    const float* min_b = (const float*)d_in[20];
    const float* mout_w= (const float*)d_in[21];
    const float* mout_b= (const float*)d_in[22];
    const float* norm_g= (const float*)d_in[23];
    const float* norm_b= (const float*)d_in[24];
    const float* out_w1= (const float*)d_in[25];
    const float* out_b1= (const float*)d_in[26];
    const float* out_w2= (const float*)d_in[27];
    const float* out_b2= (const float*)d_in[28];
    float* out = (float*)d_out;

    char* wp = (char*)d_ws;
    auto take = [&](size_t bytes) -> void* {
        void* p = (void*)wp;
        wp += (bytes + 255) & ~(size_t)255;
        return p;
    };

    float* x   = (float*)take((size_t)TOK * DM * 4);
    u16*   hbf = (u16*)  take((size_t)TOK * DM * 2);   // reused: LN out / o bf16 / flat bf16
    float* q   = (float*)take((size_t)TOK * DM * 4);
    float* k   = (float*)take((size_t)TOK * DM * 4);
    float* v   = (float*)take((size_t)TOK * DM * 4);
    float* o   = (float*)take((size_t)TOK * DM * 4);
    u16*   fbf = (u16*)  take((size_t)TOK * FF * 2);
    float* gbuf= (float*)take((size_t)BATCH * FF * 4);
    u16* wq_bf = (u16*)take((size_t)NL * DM * DM * 2);
    u16* wk_bf = (u16*)take((size_t)NL * DM * DM * 2);
    u16* wv_bf = (u16*)take((size_t)NL * DM * DM * 2);
    u16* w1_bf = (u16*)take((size_t)NL * FF * DM * 2);
    u16* w2_bf = (u16*)take((size_t)NL * DM * FF * 2);
    u16* min_bf= (u16*)take((size_t)3 * DM * DM * 2);
    u16* mout_bf=(u16*)take((size_t)DM * DM * 2);
    u16* ow1_bf= (u16*)take((size_t)FF * SEQ * DM * 2);

    auto conv = [&](const float* src, u16* dst, size_t n) {
        k_f32_to_bf16<<<(unsigned)((n + 255) / 256), 256, 0, stream>>>(src, dst, n);
    };
    conv(Wq,  wq_bf,  (size_t)NL * DM * DM);
    conv(Wk,  wk_bf,  (size_t)NL * DM * DM);
    conv(Wv,  wv_bf,  (size_t)NL * DM * DM);
    conv(W1f, w1_bf,  (size_t)NL * FF * DM);
    conv(W2f, w2_bf,  (size_t)NL * DM * FF);
    conv(min_w,  min_bf,  (size_t)3 * DM * DM);
    conv(mout_w, mout_bf, (size_t)DM * DM);
    conv(out_w1, ow1_bf,  (size_t)FF * SEQ * DM);

    // Embedding -> x
    {
        size_t n = (size_t)TOK * DM;
        k_embed<<<(unsigned)((n + 255) / 256), 256, 0, stream>>>(
            dense_x, sparse_x, dense_w, dense_b, emb, x);
    }

    const float layer_scale = 1.0f / 16.0f;            // 1/sqrt(D=256)
    const float mha_scale   = 0.1767766952966369f;     // 1/sqrt(HD=32)
    dim3 gD(TOK / 256, DM / 64);      // GEMM grid, N=256 (block covers 256 rows x 64 cols)
    dim3 gF(TOK / 256, FF / 64);      // GEMM grid, N=1024
    dim3 gA(BATCH, NH);               // attention grid

    for (int i = 0; i < NL; ++i) {
        // LN1 -> hbf
        k_layernorm_bf16<<<TOK, 256, 0, stream>>>(x, ln1_g + i * DM, ln1_b + i * DM, hbf);
        // Q/K/V projections
        k_gemm_wmma<0,0,0><<<gD, 256, 0, stream>>>(hbf, wq_bf + (size_t)i * DM * DM,
            bq + i * DM, q, nullptr, nullptr, TOK, DM, DM);
        k_gemm_wmma<0,0,0><<<gD, 256, 0, stream>>>(hbf, wk_bf + (size_t)i * DM * DM,
            bk + i * DM, k, nullptr, nullptr, TOK, DM, DM);
        k_gemm_wmma<0,0,0><<<gD, 256, 0, stream>>>(hbf, wv_bf + (size_t)i * DM * DM,
            bv + i * DM, v, nullptr, nullptr, TOK, DM, DM);
        // sigmoid attention, x += o
        k_attn_sigmoid<<<gA, 256, 0, stream>>>(q, k, v, x, layer_scale);
        // LN2 -> hbf
        k_layernorm_bf16<<<TOK, 256, 0, stream>>>(x, ln2_g + i * DM, ln2_b + i * DM, hbf);
        // FFN1: relu(h @ W1^T + b1) -> fbf (bf16)
        k_gemm_wmma<1,0,1><<<gF, 256, 0, stream>>>(hbf, w1_bf + (size_t)i * FF * DM,
            b1f + i * FF, nullptr, fbf, nullptr, TOK, FF, DM);
        // FFN2: x = x + f @ W2^T + b2
        k_gemm_wmma<0,1,0><<<gD, 256, 0, stream>>>(fbf, w2_bf + (size_t)i * DM * FF,
            b2f + i * DM, x, nullptr, x, TOK, DM, FF);
    }

    // Final MHA
    k_layernorm_bf16<<<TOK, 256, 0, stream>>>(x, norm_g, norm_b, hbf);
    k_gemm_wmma<0,0,0><<<gD, 256, 0, stream>>>(hbf, min_bf,                    min_b,          q, nullptr, nullptr, TOK, DM, DM);
    k_gemm_wmma<0,0,0><<<gD, 256, 0, stream>>>(hbf, min_bf + (size_t)DM * DM,  min_b + DM,     k, nullptr, nullptr, TOK, DM, DM);
    k_gemm_wmma<0,0,0><<<gD, 256, 0, stream>>>(hbf, min_bf + (size_t)2*DM*DM,  min_b + 2*DM,   v, nullptr, nullptr, TOK, DM, DM);
    k_attn_softmax<<<gA, 256, 0, stream>>>(q, k, v, o, mha_scale);
    // o bf16 -> hbf; x = x + o @ mha_out_w^T + b
    conv(o, hbf, (size_t)TOK * DM);
    k_gemm_wmma<0,1,0><<<gD, 256, 0, stream>>>(hbf, mout_bf, mout_b, x, nullptr, x, TOK, DM, DM);

    // flat bf16 -> hbf ; gbuf = relu(flat @ out_w1^T + out_b1)
    conv(x, hbf, (size_t)TOK * DM);
    dim3 gO(BATCH / 256, FF / 64);
    k_gemm_wmma<1,0,0><<<gO, 256, 0, stream>>>(hbf, ow1_bf, out_b1, gbuf, nullptr, nullptr,
                                               BATCH, FF, SEQ * DM);
    // out[b] = gbuf[b] . out_w2 + out_b2
    k_final_reduce<<<BATCH, 256, 0, stream>>>(gbuf, out_w2, out_b2, out);
}
